// VectorQuantizer_17025250361871
// MI455X (gfx1250) — matrix operand order repaired and checked by reading, not compiled, not run on hardware
//
#include <hip/hip_runtime.h>
#include <hip/hip_bf16.h>

// ---------------------------------------------------------------------------
// VQ-VAE vector quantizer for MI455X (gfx1250, wave32, WMMA).
// Distance GEMM done as bf16x3 split-precision WMMA (hi*hi + hi*lo + lo*hi),
// giving ~fp32 accuracy at bf16 matrix-core rates (memory-roof bound).
// Codebook slabs staged to LDS with gfx1250 async global->LDS copies
// (ASYNCcnt path) -- builtin confirmed present on this toolchain.
// ---------------------------------------------------------------------------

typedef __attribute__((ext_vector_type(16))) __bf16 v16bf;
typedef __attribute__((ext_vector_type(8)))  float  v8f;

#define N_CODES     1024
#define E_DIM       256
#define NROWS       32768        // 32 * 32 * 32  (B*H*W)
#define HW          1024         // 32 * 32
#define TOTAL_ELEMS 8388608      // 32 * 256 * 32 * 32
#define BETA_F      0.25f

#if __has_builtin(__builtin_amdgcn_global_load_async_to_lds_b32) && \
    __has_builtin(__builtin_amdgcn_s_wait_asynccnt)
#define HAVE_ASYNC_LDS 1
#endif

__device__ __forceinline__ unsigned short f2bf_rne(float f) {
    unsigned int u = __float_as_uint(f);
    u += 0x7FFFu + ((u >> 16) & 1u);          // round-to-nearest-even
    return (unsigned short)(u >> 16);
}
__device__ __forceinline__ float bf2f(unsigned short h) {
    return __uint_as_float(((unsigned int)h) << 16);
}

// ---------------------------------------------------------------------------
// Prep: codebook -> bf16 hi/lo split, ||e||^2 per code, zero accumulators.
// One block per code.
// ---------------------------------------------------------------------------
__global__ __launch_bounds__(256) void vq_prep(
        const float* __restrict__ weight,
        unsigned short* __restrict__ whi, unsigned short* __restrict__ wlo,
        float* __restrict__ wsq, unsigned int* __restrict__ counts,
        float* __restrict__ sumsq) {
    __shared__ float red[256];
    const int j = blockIdx.x;
    const int t = threadIdx.x;
    float x = weight[j * E_DIM + t];
    unsigned short h = f2bf_rne(x);
    whi[j * E_DIM + t] = h;
    wlo[j * E_DIM + t] = f2bf_rne(x - bf2f(h));
    red[t] = x * x;
    __syncthreads();
    for (int s = 128; s > 0; s >>= 1) {
        if (t < s) red[t] += red[t + s];
        __syncthreads();
    }
    if (t == 0) {
        wsq[j] = red[0];
        counts[j] = 0u;
        if (j == 0) *sumsq = 0.0f;
    }
}

// ---------------------------------------------------------------------------
// Argmin GEMM: each block = 4 waves, 64 rows; loops over 16 blocks of 64 codes.
// A (z rows) and B (codes) staged in LDS as bf16 hi/lo; per-wave 16x64 f32
// register tile; 3x v_wmma_f32_16x16x32_bf16 per 16x16x32 sub-tile.
// Score = ||e||^2 - 2 z.e   (||z||^2 is row-constant; argmin-invariant).
// ---------------------------------------------------------------------------
__global__ __launch_bounds__(128) void vq_argmin(
        const float* __restrict__ z,
        const unsigned short* __restrict__ whi,
        const unsigned short* __restrict__ wlo,
        const float* __restrict__ wsq,
        int* __restrict__ idx_out) {
    __shared__ unsigned short a_hi[64][E_DIM];   // 32 KB
    __shared__ unsigned short a_lo[64][E_DIM];   // 32 KB
    __shared__ unsigned short b_hi[64][E_DIM];   // 32 KB
    __shared__ unsigned short b_lo[64][E_DIM];   // 32 KB

    const int rowBase = blockIdx.x * 64;
    const int bImg    = rowBase >> 10;           // image index (b): 1024 rows per image
    const int rem0    = rowBase & 1023;          // h*32+w offset, contiguous in memory
    const int tid     = threadIdx.x;

    // Stage A: z[b][c][h][w] with (h,w) contiguous -> coalesced per channel c.
    for (int it = 0; it < 128; ++it) {
        int lin = it * 128 + tid;                // 64 rows * 256 channels
        int i = lin & 63;
        int c = lin >> 6;
        float x = z[((bImg * E_DIM + c) << 10) + rem0 + i];
        unsigned short h = f2bf_rne(x);
        a_hi[i][c] = h;
        a_lo[i][c] = f2bf_rne(x - bf2f(h));
    }

    const int wave = tid >> 5;
    const int lane = tid & 31;
    const int lm   = lane & 15;
    const int lh   = lane >> 4;
    const int rowL = wave * 16 + lm;             // A-fragment row for this lane

    float best[8];
    int   bidx[8];
    #pragma unroll
    for (int r = 0; r < 8; ++r) { best[r] = 3.4e38f; bidx[r] = 0; }

    for (int nblk = 0; nblk < 16; ++nblk) {
        const int cbase = nblk * 64;
        __syncthreads();
        // Stage 64-code slab of codebook (hi/lo) into LDS.
        {
            const unsigned int* sh = (const unsigned int*)(whi + cbase * E_DIM);
            const unsigned int* sl = (const unsigned int*)(wlo + cbase * E_DIM);
            unsigned int* dh = (unsigned int*)&b_hi[0][0];
            unsigned int* dl = (unsigned int*)&b_lo[0][0];
#ifdef HAVE_ASYNC_LDS
            // gfx1250 async global->LDS copy: no VGPR round-trip, ASYNCcnt.
            typedef __attribute__((address_space(1))) int GiT;
            typedef __attribute__((address_space(3))) int LiT;
            for (int t2 = tid; t2 < 64 * (E_DIM / 2); t2 += 128) {
                // Flat LDS address low 32 bits == LDS offset (aperture layout).
                GiT* gsh = (GiT*)(unsigned long long)(sh + t2);
                GiT* gsl = (GiT*)(unsigned long long)(sl + t2);
                LiT* ldh = (LiT*)(unsigned int)(unsigned long long)(dh + t2);
                LiT* ldl = (LiT*)(unsigned int)(unsigned long long)(dl + t2);
                __builtin_amdgcn_global_load_async_to_lds_b32(gsh, ldh, 0, 0);
                __builtin_amdgcn_global_load_async_to_lds_b32(gsl, ldl, 0, 0);
            }
            __builtin_amdgcn_s_wait_asynccnt(0);
#else
            for (int t2 = tid; t2 < 64 * (E_DIM / 2); t2 += 128) {
                dh[t2] = sh[t2];
                dl[t2] = sl[t2];
            }
#endif
        }
        __syncthreads();

        // Prefetch next slab into cache while we compute on this one.
        if (nblk < 15) {
            const char* nh = (const char*)(whi + (cbase + 64) * E_DIM);
            const char* nl = (const char*)(wlo + (cbase + 64) * E_DIM);
            // 32KB slab = 128 x 256B cachelines; 128 threads -> 2 lines each.
            __builtin_prefetch(nh + tid * 256, 0, 1);
            __builtin_prefetch(nh + 32768 - 256 - tid * 256, 0, 1);
            __builtin_prefetch(nl + tid * 256, 0, 1);
            __builtin_prefetch(nl + 32768 - 256 - tid * 256, 0, 1);
        }

        v8f acc[4] = { v8f{}, v8f{}, v8f{}, v8f{} };

        for (int ks = 0; ks < 8; ++ks) {         // K: 8 slices of 32
            union { unsigned int u[8]; v16bf v; } Ah, Al;
            // 16-bit A 16x32 layout: VGPR v holds K pair
            //   kb = ks*32 + (v<4 ? 0 : 16) + (lane/16)*8 + (v%4)*2
            #pragma unroll
            for (int v = 0; v < 8; ++v) {
                int kb = ks * 32 + ((v < 4) ? 0 : 16) + lh * 8 + (v & 3) * 2;
                Ah.u[v] = *(const unsigned int*)&a_hi[rowL][kb];
                Al.u[v] = *(const unsigned int*)&a_lo[rowL][kb];
            }
            #pragma unroll
            for (int t3 = 0; t3 < 4; ++t3) {     // 4 N-tiles of 16 codes
                const int cl = t3 * 16 + lm;     // B column = code (local)
                union { unsigned int u[8]; v16bf v; } Bh, Bl;
                // B 32x16 layout: same K order as A,
                // VGPR v holds K = (v<4 ? 0 : 16) + half*8 + (v%4)*2, +1
                #pragma unroll
                for (int v = 0; v < 8; ++v) {
                    int ko = ks * 32 + ((v < 4) ? 0 : 16) + lh * 8 + (v & 3) * 2;
                    Bh.u[v] = *(const unsigned int*)&b_hi[cl][ko];
                    Bl.u[v] = *(const unsigned int*)&b_lo[cl][ko];
                }
                // bf16x3: hi*hi + hi*lo + lo*hi chained into same accumulator
                acc[t3] = __builtin_amdgcn_wmma_f32_16x16x32_bf16(
                    false, Ah.v, false, Bh.v, (short)0, acc[t3], false, false);
                acc[t3] = __builtin_amdgcn_wmma_f32_16x16x32_bf16(
                    false, Ah.v, false, Bl.v, (short)0, acc[t3], false, false);
                acc[t3] = __builtin_amdgcn_wmma_f32_16x16x32_bf16(
                    false, Al.v, false, Bh.v, (short)0, acc[t3], false, false);
            }
        }

        // Running argmin. C/D layout: VGPR r, lane l -> M = r + 8*(l/16), N = l%16.
        #pragma unroll
        for (int t3 = 0; t3 < 4; ++t3) {
            const int code = cbase + t3 * 16 + lm;
            const float wq = wsq[code];
            #pragma unroll
            for (int r = 0; r < 8; ++r) {
                float s = wq - 2.0f * acc[t3][r];
                if (s < best[r]) { best[r] = s; bidx[r] = code; }
            }
        }
    }

    // Cross-lane reduction within each 16-lane half (first-index tie-break).
    #pragma unroll
    for (int r = 0; r < 8; ++r) {
        float v = best[r];
        int   bi = bidx[r];
        #pragma unroll
        for (int off = 1; off < 16; off <<= 1) {
            float ov = __shfl_xor(v, off, 32);
            int   oi = __shfl_xor(bi, off, 32);
            if (ov < v || (ov == v && oi < bi)) { v = ov; bi = oi; }
        }
        if (lm == 0) idx_out[rowBase + wave * 16 + r + 8 * lh] = bi;
    }
}

// ---------------------------------------------------------------------------
// Gather z_q, write straight-through output z + (w[idx] - z), accumulate
// sum((z_q - z)^2) with one atomic per block.
// ---------------------------------------------------------------------------
__global__ __launch_bounds__(256) void vq_gather(
        const float* __restrict__ z, const float* __restrict__ weight,
        const int* __restrict__ idx, float* __restrict__ zq,
        float* __restrict__ sumsq) {
    __shared__ float red[256];
    float local = 0.0f;
    const int gsize = gridDim.x * blockDim.x;
    for (int e = blockIdx.x * blockDim.x + threadIdx.x; e < TOTAL_ELEMS; e += gsize) {
        int p  = e & 1023;           // h*32 + w
        int c  = (e >> 10) & 255;    // channel
        int bb = e >> 18;            // image
        int j  = idx[bb * HW + p];
        float wv = weight[j * E_DIM + c];
        float zv = z[e];
        float d  = wv - zv;
        local += d * d;
        zq[e] = zv + d;              // matches zp + stop_grad(z_q - zp) bitwise
    }
    red[threadIdx.x] = local;
    __syncthreads();
    for (int s = 128; s > 0; s >>= 1) {
        if (threadIdx.x < s) red[threadIdx.x] += red[threadIdx.x + s];
        __syncthreads();
    }
    if (threadIdx.x == 0) atomicAdd(sumsq, red[0]);
}

// ---------------------------------------------------------------------------
// Codebook usage histogram.
// ---------------------------------------------------------------------------
__global__ __launch_bounds__(256) void vq_counts(
        const int* __restrict__ idx, unsigned int* __restrict__ counts) {
    int t = blockIdx.x * blockDim.x + threadIdx.x;
    if (t < NROWS) atomicAdd(&counts[idx[t]], 1u);
}

// ---------------------------------------------------------------------------
// Finalize: loss = (1+beta) * sumsq / nelem ; perplexity = exp(-sum e*log e).
// ---------------------------------------------------------------------------
__global__ __launch_bounds__(256) void vq_finalize(
        const float* __restrict__ sumsq, const unsigned int* __restrict__ counts,
        float* __restrict__ loss_out, float* __restrict__ perp_out) {
    __shared__ float red[256];
    const int t = threadIdx.x;
    float acc = 0.0f;
    for (int j = t; j < N_CODES; j += 256) {
        float em = (float)counts[j] / (float)NROWS;
        acc += em * logf(em + 1e-10f);
    }
    red[t] = acc;
    __syncthreads();
    for (int s = 128; s > 0; s >>= 1) {
        if (t < s) red[t] += red[t + s];
        __syncthreads();
    }
    if (t == 0) {
        *perp_out = expf(-red[0]);
        *loss_out = (1.0f + BETA_F) * (*sumsq) / (float)TOTAL_ELEMS;
    }
}

// ---------------------------------------------------------------------------
extern "C" void kernel_launch(void* const* d_in, const int* in_sizes, int n_in,
                              void* d_out, int out_size, void* d_ws, size_t ws_size,
                              hipStream_t stream) {
    const float* z      = (const float*)d_in[0];   // (32,256,32,32) f32
    const float* weight = (const float*)d_in[1];   // (1024,256) f32

    float* out      = (float*)d_out;
    float* loss_out = out;                          // [0]
    float* zq       = out + 1;                      // [1 .. 8388608]
    float* perp_out = out + 1 + TOTAL_ELEMS;        // [8388609]
    int*   idx_out  = (int*)(out + 2 + TOTAL_ELEMS);// [8388610 .. 8421377]

    char* ws = (char*)d_ws;
    unsigned short* whi   = (unsigned short*)(ws);            // 512 KB
    unsigned short* wlo   = (unsigned short*)(ws + 524288);   // 512 KB
    float*          wsq   = (float*)(ws + 1048576);           // 4 KB
    unsigned int*   cnts  = (unsigned int*)(ws + 1052672);    // 4 KB
    float*          sumsq = (float*)(ws + 1056768);           // 4 B

    vq_prep   <<<N_CODES,     256, 0, stream>>>(weight, whi, wlo, wsq, cnts, sumsq);
    vq_argmin <<<NROWS / 64,  128, 0, stream>>>(z, whi, wlo, wsq, idx_out);
    vq_gather <<<4096,        256, 0, stream>>>(z, weight, idx_out, zq, sumsq);
    vq_counts <<<NROWS / 256, 256, 0, stream>>>(idx_out, cnts);
    vq_finalize<<<1,          256, 0, stream>>>(sumsq, cnts, loss_out, perp_out);
}
